// Head_66219805769747
// MI455X (gfx1250) — matrix operand order repaired and checked by reading, not compiled, not run on hardware
//
#include <hip/hip_runtime.h>
#include <hip/hip_bf16.h>
#include <stdint.h>

// Problem constants (from reference): B=16, T=2048, C=1024, H=64
constexpr int Bn = 16;
constexpr int Tn = 2048;
constexpr int Cn = 1024;
constexpr int Hn = 64;
constexpr int BT = Bn * Tn;
constexpr float SCALE = 0.125f;  // 1/sqrt(64)

typedef __attribute__((ext_vector_type(16))) __bf16 v16bf;
typedef __attribute__((ext_vector_type(8)))  float  v8f;
typedef __attribute__((ext_vector_type(4)))  uint32_t u32x4;
typedef __attribute__((ext_vector_type(8)))  int      i32x8;
typedef __attribute__((ext_vector_type(4)))  int      i32x4;

#define WMMA_BF16(a, b, c) \
  __builtin_amdgcn_wmma_f32_16x16x32_bf16(false, (a), false, (b), (short)0, (c), false, false)

#if defined(__gfx1250__) && __has_builtin(__builtin_amdgcn_tensor_load_to_lds) && \
    __has_builtin(__builtin_amdgcn_s_wait_tensorcnt)
#define USE_TDM 1
#else
#define USE_TDM 0
#endif

#if USE_TDM
// One TDM load of a 2-D bf16 tile (64 cols x tile_rows rows, memory row stride
// 64 elements) into LDS, padding 4 dwords after every 32 dwords stored ->
// effective LDS row stride of 72 bf16, matching the fragment loaders.
__device__ __forceinline__ void tdm_load_tile(uint32_t lds_off, const void* gptr,
                                              int tile_rows) {
  uint64_t ga = (uint64_t)(uintptr_t)gptr;
  u32x4 g0;
  g0[0] = 1u;                                   // count=1, user descriptor
  g0[1] = lds_off;                              // lds_addr (bytes)
  g0[2] = (uint32_t)ga;                         // global_addr[31:0]
  g0[3] = (uint32_t)((ga >> 32) & 0x1FFFFFFu) | (2u << 30);  // addr[56:32] | type=2
  i32x8 g1;
  g1[0] = (1 << 16) |                           // data_size = 2 bytes
          (1 << 20) |                           // pad_enable
          (4 << 22) |                           // pad_interval = 32 dwords
          (3 << 25);                            // pad_amount  = 4 dwords
  g1[1] = (int)((uint32_t)Hn << 16);            // tensor_dim0 = 64
  g1[2] = (int)((uint32_t)(BT & 0xFFFF) << 16); // tensor_dim1 low16
  g1[3] = (int)(((uint32_t)BT >> 16) & 0xFFFF) |
          (int)((uint32_t)Hn << 16);            // tensor_dim1 hi | tile_dim0 = 64
  g1[4] = tile_rows;                            // tile_dim1 | tile_dim2 = 0
  g1[5] = Hn;                                   // tensor_dim0_stride = 64
  g1[6] = 0;
  g1[7] = 0;
  i32x4 z4 = {};
#if __clang_major__ >= 23
  i32x8 z8 = {};
  __builtin_amdgcn_tensor_load_to_lds(g0, g1, z4, z4, z8, 0);
#else
  __builtin_amdgcn_tensor_load_to_lds(g0, g1, z4, z4, 0);
#endif
}
#endif

// ---------------------------------------------------------------------------
// Kernel 1: QKV projection, double-buffered.  y = x @ W^T, stored as bf16.
// grid = (BT/128, 3): blockIdx.y selects (q,k,v); block = 256 threads (8 waves).
// ---------------------------------------------------------------------------
__global__ __launch_bounds__(256)
void qkv_proj_kernel(const float* __restrict__ x,
                     const float* __restrict__ Wk,
                     const float* __restrict__ Wq,
                     const float* __restrict__ Wv,
                     __bf16* __restrict__ qws,
                     __bf16* __restrict__ kws,
                     __bf16* __restrict__ vws) {
  __shared__ __align__(16) __bf16 xs[2][128][40];   // double-buffered x tile
  __shared__ __align__(16) __bf16 wsm[2][64][40];   // double-buffered W tile

  const float* W;
  __bf16* out;
  if (blockIdx.y == 0)      { W = Wq; out = qws; }
  else if (blockIdx.y == 1) { W = Wk; out = kws; }
  else                      { W = Wv; out = vws; }

  const int tid  = threadIdx.x;
  const int lane = tid & 31;
  const int l15  = lane & 15;
  const int hf   = lane >> 4;
  const int wv   = tid >> 5;
  const int m0   = wv * 16;
  const size_t rbase = (size_t)blockIdx.x * 128;

  const float4* x4 = (const float4*)x;
  const float4* w4 = (const float4*)W;

  auto stage = [&](int buf, int k0) {
#pragma unroll
    for (int t = 0; t < 4; ++t) {
      int idx = tid + t * 256;
      int r = idx >> 3, c4 = idx & 7;
      float4 f = x4[(rbase + r) * (Cn / 4) + (k0 >> 2) + c4];
      xs[buf][r][c4 * 4 + 0] = (__bf16)f.x;
      xs[buf][r][c4 * 4 + 1] = (__bf16)f.y;
      xs[buf][r][c4 * 4 + 2] = (__bf16)f.z;
      xs[buf][r][c4 * 4 + 3] = (__bf16)f.w;
    }
#pragma unroll
    for (int t = 0; t < 2; ++t) {
      int idx = tid + t * 256;
      int r = idx >> 3, c4 = idx & 7;
      float4 f = w4[(size_t)r * (Cn / 4) + (k0 >> 2) + c4];
      wsm[buf][r][c4 * 4 + 0] = (__bf16)f.x;
      wsm[buf][r][c4 * 4 + 1] = (__bf16)f.y;
      wsm[buf][r][c4 * 4 + 2] = (__bf16)f.z;
      wsm[buf][r][c4 * 4 + 3] = (__bf16)f.w;
    }
  };

  v8f acc[4];
#pragma unroll
  for (int t = 0; t < 4; ++t) acc[t] = v8f{};

  constexpr int NIT = Cn / 32;   // 32 K-iterations
  stage(0, 0);
  __syncthreads();

  for (int it = 0; it < NIT; ++it) {
    const int cur = it & 1;
    if (it + 1 < NIT) stage(cur ^ 1, (it + 1) * 32);  // prefetch next tile

    v16bf a;
    {
      int r  = m0 + l15;
      int kb = hf ? 16 : 0;
#pragma unroll
      for (int e = 0; e < 8; ++e) {
        a[e]     = xs[cur][r][kb + e];
        a[8 + e] = xs[cur][r][8 + kb + e];
      }
    }
#pragma unroll
    for (int t = 0; t < 4; ++t) {
      v16bf bfr;
      int n  = t * 16 + l15;
      int kk = hf ? 16 : 0;
#pragma unroll
      for (int e = 0; e < 16; ++e) bfr[e] = wsm[cur][n][kk + e];
      acc[t] = WMMA_BF16(a, bfr, acc[t]);
    }
    __syncthreads();   // single barrier: next iter may overwrite buf `cur`
  }

#pragma unroll
  for (int t = 0; t < 4; ++t) {
    int n = t * 16 + l15;
#pragma unroll
    for (int j = 0; j < 8; ++j) {
      size_t row = rbase + m0 + hf * 8 + j;
      out[row * Hn + n] = (__bf16)acc[t][j];
    }
  }
}

// ---------------------------------------------------------------------------
// Kernel 2: fused causal flash attention, double-buffered K/V staging.
// grid = (T/128, B); block = 256 (8 waves); each wave owns 16 query rows.
// ---------------------------------------------------------------------------
__global__ __launch_bounds__(256)
void flash_attn_kernel(const __bf16* __restrict__ qws,
                       const __bf16* __restrict__ kws,
                       const __bf16* __restrict__ vws,
                       float* __restrict__ out) {
  __shared__ __align__(16) __bf16 qs[128][72];        // Q tile (padded rows)
  __shared__ __align__(16) __bf16 ksm[2][64][72];     // K tiles, row-major
  __shared__ __align__(16) __bf16 vsmT[2][64][72];    // V tiles, TRANSPOSED [h][key]
  __shared__ __align__(16) __bf16 ps[8][16][72];      // per-wave P scratch 16x64

  const int tid  = threadIdx.x;
  const int lane = tid & 31;
  const int l15  = lane & 15;
  const int hf   = lane >> 4;
  const int wv   = tid >> 5;
  const int b     = blockIdx.y;
  const int qbase = blockIdx.x * 128;
  const size_t qrow0 = (size_t)b * Tn + qbase;

  // Stage a K (TDM or manual) and transposed-V tile pair into buffer `buf`.
  auto stageKV = [&](int buf, int k0) {
#if USE_TDM
    if (tid < 32) {   // wave 0 fires the DMA; no wait here (overlaps compute)
      tdm_load_tile((uint32_t)(uintptr_t)&ksm[buf][0][0],
                    kws + ((size_t)b * Tn + k0) * Hn, 64);
    }
#else
    {
      const uint32_t* ksrc = (const uint32_t*)(kws + ((size_t)b * Tn + k0) * Hn);
#pragma unroll
      for (int t = 0; t < 8; ++t) {
        int idx = tid + t * 256;
        int r = idx >> 5, c = idx & 31;
        *(uint32_t*)&ksm[buf][r][c * 2] = ksrc[(size_t)r * 32 + c];
      }
    }
#endif
    const uint32_t* vsrc = (const uint32_t*)(vws + ((size_t)b * Tn + k0) * Hn);
#pragma unroll
    for (int t = 0; t < 8; ++t) {
      int idx = tid + t * 256;
      int r = idx >> 5, c = idx & 31;        // key r, dword c -> h = 2c, 2c+1
      uint32_t d = vsrc[(size_t)r * 32 + c];
      vsmT[buf][2 * c][r]     = __builtin_bit_cast(__bf16, (uint16_t)(d & 0xFFFF));
      vsmT[buf][2 * c + 1][r] = __builtin_bit_cast(__bf16, (uint16_t)(d >> 16));
    }
  };

  // ---- Prologue: Q tile + first K/V tile ----
#if USE_TDM
  if (tid < 32) {
    tdm_load_tile((uint32_t)(uintptr_t)&qs[0][0], qws + qrow0 * Hn, 128);
  }
#else
  {
    const uint32_t* src = (const uint32_t*)(qws + qrow0 * Hn);
#pragma unroll
    for (int t = 0; t < 16; ++t) {
      int idx = tid + t * 256;
      int r = idx >> 5, c = idx & 31;
      *(uint32_t*)&qs[r][c * 2] = src[(size_t)r * 32 + c];
    }
  }
#endif
  stageKV(0, 0);
#if USE_TDM
  if (tid < 32) __builtin_amdgcn_s_wait_tensorcnt(0);  // Q + K0 both landed
#endif
  __syncthreads();

  v16bf qa0, qa1;
  {
    int r  = wv * 16 + l15;
    int kb = hf ? 16 : 0;
#pragma unroll
    for (int e = 0; e < 8; ++e) {
      qa0[e]     = qs[r][kb + e];
      qa0[8 + e] = qs[r][8 + kb + e];
      qa1[e]     = qs[r][32 + kb + e];
      qa1[8 + e] = qs[r][40 + kb + e];
    }
  }

  float mrow[8], lrow[8];
  v8f o[4];
#pragma unroll
  for (int j = 0; j < 8; ++j) { mrow[j] = -3.0e38f; lrow[j] = 0.0f; }
#pragma unroll
  for (int t = 0; t < 4; ++t) o[t] = v8f{};

  const int kend = qbase + 128;  // causal trim
  int cur = 0;

  for (int k0 = 0; k0 < kend; k0 += 64) {
    // ---- Fire next tile's staging first: DMA/loads overlap this tile's math ----
    if (k0 + 64 < kend) stageKV(cur ^ 1, k0 + 64);

    // ---- Scores S = Q K^T (4 key sub-tiles of 16) ----
    v8f s[4];
#pragma unroll
    for (int t = 0; t < 4; ++t) {
      v16bf kb0, kb1;
      int n  = t * 16 + l15;
      int kk = hf ? 16 : 0;
#pragma unroll
      for (int e = 0; e < 16; ++e) {
        kb0[e] = ksm[cur][n][kk + e];          // B[K=h][N=key] = K^T (contiguous)
        kb1[e] = ksm[cur][n][32 + kk + e];
      }
      v8f a0 = v8f{};
      a0 = WMMA_BF16(qa0, kb0, a0);
      a0 = WMMA_BF16(qa1, kb1, a0);
      s[t] = a0;
    }

    // ---- Scale + causal mask ----
#pragma unroll
    for (int t = 0; t < 4; ++t) {
      int kg = k0 + t * 16 + l15;
#pragma unroll
      for (int j = 0; j < 8; ++j) {
        int qg = qbase + wv * 16 + hf * 8 + j;
        float sv = s[t][j] * SCALE;
        s[t][j] = (kg <= qg) ? sv : -3.0e38f;
      }
    }

    // ---- Online softmax update ----
#pragma unroll
    for (int j = 0; j < 8; ++j) {
      float mn = mrow[j];
#pragma unroll
      for (int t = 0; t < 4; ++t) mn = fmaxf(mn, s[t][j]);
      mn = fmaxf(mn, __shfl_xor(mn, 1, 32));
      mn = fmaxf(mn, __shfl_xor(mn, 2, 32));
      mn = fmaxf(mn, __shfl_xor(mn, 4, 32));
      mn = fmaxf(mn, __shfl_xor(mn, 8, 32));
      float alpha = __expf(mrow[j] - mn);
      mrow[j] = mn;
      float rs = 0.0f;
#pragma unroll
      for (int t = 0; t < 4; ++t) {
        float p = __expf(s[t][j] - mn);
        s[t][j] = p;
        rs += p;
      }
      rs += __shfl_xor(rs, 1, 32);
      rs += __shfl_xor(rs, 2, 32);
      rs += __shfl_xor(rs, 4, 32);
      rs += __shfl_xor(rs, 8, 32);
      lrow[j] = lrow[j] * alpha + rs;
#pragma unroll
      for (int t = 0; t < 4; ++t) o[t][j] *= alpha;
    }

    // ---- Spill P (C layout) to per-wave LDS, re-read in A layout ----
#pragma unroll
    for (int t = 0; t < 4; ++t) {
#pragma unroll
      for (int j = 0; j < 8; ++j) {
        ps[wv][hf * 8 + j][t * 16 + l15] = (__bf16)s[t][j];
      }
    }

    v16bf pa0, pa1;
    {
      int r  = l15;
      int kb = hf ? 16 : 0;
#pragma unroll
      for (int e = 0; e < 8; ++e) {
        pa0[e]     = ps[wv][r][kb + e];
        pa0[8 + e] = ps[wv][r][8 + kb + e];
        pa1[e]     = ps[wv][r][32 + kb + e];
        pa1[8 + e] = ps[wv][r][40 + kb + e];
      }
    }

    // ---- O += P V  (B[K=key][N=h]; vsmT makes these contiguous b128 reads) ----
#pragma unroll
    for (int t = 0; t < 4; ++t) {
      v16bf vb0, vb1;
      int n  = t * 16 + l15;
      int kk = hf ? 16 : 0;
#pragma unroll
      for (int e = 0; e < 16; ++e) {
        vb0[e] = vsmT[cur][n][kk + e];
        vb1[e] = vsmT[cur][n][32 + kk + e];
      }
      o[t] = WMMA_BF16(pa0, vb0, o[t]);
      o[t] = WMMA_BF16(pa1, vb1, o[t]);
    }

    // ---- Close the pipeline stage: DMA must have landed before next iter ----
#if USE_TDM
    if (tid < 32) __builtin_amdgcn_s_wait_tensorcnt(0);
#endif
    __syncthreads();
    cur ^= 1;
  }

  // ---- Epilogue: normalize, write fp32 output [B,T,H] ----
#pragma unroll
  for (int t = 0; t < 4; ++t) {
    int n = t * 16 + l15;
#pragma unroll
    for (int j = 0; j < 8; ++j) {
      size_t row = qrow0 + wv * 16 + hf * 8 + j;
      out[row * Hn + n] = o[t][j] / lrow[j];
    }
  }
}

// ---------------------------------------------------------------------------
extern "C" void kernel_launch(void* const* d_in, const int* in_sizes, int n_in,
                              void* d_out, int out_size, void* d_ws, size_t ws_size,
                              hipStream_t stream) {
  const float* x  = (const float*)d_in[0];
  const float* Wk = (const float*)d_in[1];
  const float* Wq = (const float*)d_in[2];
  const float* Wv = (const float*)d_in[3];
  float* out = (float*)d_out;

  __bf16* qws = (__bf16*)d_ws;                 // bf16 q/k/v workspace, 12 MB
  __bf16* kws = qws + (size_t)BT * Hn;
  __bf16* vws = kws + (size_t)BT * Hn;

  dim3 g1(BT / 128, 3), b1(256);
  qkv_proj_kernel<<<g1, b1, 0, stream>>>(x, Wk, Wq, Wv, qws, kws, vws);

  dim3 g2(Tn / 128, Bn), b2(256);
  flash_attn_kernel<<<g2, b2, 0, stream>>>(qws, kws, vws, out);
}
